// Attention_63290638074308
// MI455X (gfx1250) — matrix operand order repaired and checked by reading, not compiled, not run on hardware
//
#include <hip/hip_runtime.h>

typedef __attribute__((ext_vector_type(16))) _Float16 v16h;
typedef __attribute__((ext_vector_type(8)))  float    v8f;

#define EMBED    1024
#define HEADS    16
#define HEAD_DIM 64
#define NUM_SEQS 8
#define SEQ_MAX  1024
#define TOTAL    (NUM_SEQS * SEQ_MAX)

#define WMMA_F16(A, B, C) \
  __builtin_amdgcn_wmma_f32_16x16x32_f16(false, (A), false, (B), (short)0, (C), false, false)

// ---------------- fragment loaders (ISA 7.12.2 layouts, wave32) ----------------

// A-matrix 16x32 f16: lane&15 = M row; lanes 0-15 hold K {0..7,16..23}, lanes 16-31 hold K {8..15,24..31}
__device__ __forceinline__ v16h load_a_frag(const _Float16* base, int stride) {
  const int l    = threadIdx.x & 31;
  const int row  = l & 15;
  const int koff = (l < 16) ? 0 : 8;
  const _Float16* p = base + row * stride;
  v16h a;
#pragma unroll
  for (int e = 0; e < 16; ++e) {
    int k = ((e < 8) ? e : (e + 8)) + koff;
    a[e] = p[k];
  }
  return a;
}

// B-matrix 32x16 f16 from [n][k]-major storage (i.e. W row-major for C = A*W^T)
__device__ __forceinline__ v16h load_b_frag_nk(const _Float16* base, int stride) {
  const int l     = threadIdx.x & 31;
  const int n     = l & 15;
  const int kbase = (l < 16) ? 0 : 8;
  const _Float16* p = base + n * stride + kbase;
  v16h b;
#pragma unroll
  for (int e = 0; e < 16; ++e) b[e] = p[(e < 8) ? e : (e + 8)];
  return b;
}

// B-matrix 32x16 f16 from [k][n]-major storage (V matrix in P*V)
__device__ __forceinline__ v16h load_b_frag_kn(const _Float16* base, int stride) {
  const int l     = threadIdx.x & 31;
  const int n     = l & 15;
  const int kbase = (l < 16) ? 0 : 8;
  v16h b;
#pragma unroll
  for (int e = 0; e < 16; ++e) b[e] = base[(kbase + ((e < 8) ? e : (e + 8))) * stride + n];
  return b;
}

// ---------------- CDNA5 async global->LDS (ASYNCcnt path, no VGPR round trip) -----

__device__ __forceinline__ void async_ld_lds_b128(const _Float16* gptr, const _Float16* ldsptr) {
  unsigned lp = (unsigned)(unsigned long long)(const void*)ldsptr;  // low 32 bits = LDS offset
  asm volatile("global_load_async_to_lds_b128 %0, %1, off" :: "v"(lp), "v"(gptr) : "memory");
}

// ---------------- f32 -> f16 convert ----------------

__global__ void cvt_f32_to_f16(const float* __restrict__ in, _Float16* __restrict__ out, int n) {
  int i = blockIdx.x * blockDim.x + threadIdx.x;
  if (i < n) out[i] = (_Float16)in[i];
}

// ---------------- GEMM: out[M,N] = A[M,K] * W[N,K]^T + bias ----------------
// block tile 128x128, BK=32, 256 threads = 8 waves (4 in M, 2 in N), 32x64 per wave.
// Double-buffered LDS fed by global_load_async_to_lds_b128, last stage peeled so the
// steady-state loop is branch-free (unconditional issue + s_wait_asynccnt 4).

#define BM  128
#define BN  128
#define BK  32
#define LSA 48  // LDS row stride in halves (96B, 16B-aligned rows)

// stage a 128-row x 32-half tile: 512 b128 chunks, 2 per thread
__device__ __forceinline__ void stage_tile_async(const _Float16* __restrict__ g, int gstride,
                                                 _Float16* lds) {
  const int t = threadIdx.x;
#pragma unroll
  for (int i = 0; i < 2; ++i) {
    int c   = t + i * 256;   // 0..511
    int row = c >> 2;
    int off = (c & 3) * 8;   // halves: 0,8,16,24
    async_ld_lds_b128(g + (size_t)row * gstride + off, lds + row * LSA + off);
  }
}

__global__ __launch_bounds__(256)
void gemm_nt_kernel(const _Float16* __restrict__ A, const _Float16* __restrict__ W,
                    const float* __restrict__ bias, _Float16* __restrict__ outh,
                    float* __restrict__ outf, int M, int N, int K) {
  __shared__ _Float16 As[2][BM * LSA];
  __shared__ _Float16 Bs[2][BN * LSA];

  const int t   = threadIdx.x;
  const int wid = t >> 5;
  const int wm  = wid & 3;   // 4 waves in M
  const int wn  = wid >> 2;  // 2 waves in N
  const int bm0 = blockIdx.x * BM;
  const int bn0 = blockIdx.y * BN;
  const _Float16* Ab = A + (size_t)bm0 * K;
  const _Float16* Wb = W + (size_t)bn0 * K;

  v8f acc[2][4];
#pragma unroll
  for (int i = 0; i < 2; ++i)
#pragma unroll
    for (int j = 0; j < 4; ++j) acc[i][j] = (v8f){0.f, 0.f, 0.f, 0.f, 0.f, 0.f, 0.f, 0.f};

  auto compute_stage = [&](int cur) {
    v16h af0 = load_a_frag(&As[cur][(wm * 32 + 0) * LSA], LSA);
    v16h af1 = load_a_frag(&As[cur][(wm * 32 + 16) * LSA], LSA);
    v16h bf[4];
#pragma unroll
    for (int ni = 0; ni < 4; ++ni)
      bf[ni] = load_b_frag_nk(&Bs[cur][(wn * 64 + ni * 16) * LSA], LSA);
#pragma unroll
    for (int ni = 0; ni < 4; ++ni) {
      acc[0][ni] = WMMA_F16(af0, bf[ni], acc[0][ni]);
      acc[1][ni] = WMMA_F16(af1, bf[ni], acc[1][ni]);
    }
  };

  // prologue: stage 0
  stage_tile_async(Ab, K, As[0]);
  stage_tile_async(Wb, K, Bs[0]);

  const int KT = K / BK;
  // steady state: branch-free body (issue next, partial async wait, compute current)
  for (int kt = 0; kt < KT - 1; ++kt) {
    const int cur = kt & 1;
    stage_tile_async(Ab + (size_t)(kt + 1) * BK, K, As[cur ^ 1]);
    stage_tile_async(Wb + (size_t)(kt + 1) * BK, K, Bs[cur ^ 1]);
    asm volatile("s_wait_asynccnt 0x4" ::: "memory");  // current stage's 4 ops landed
    __syncthreads();
    compute_stage(cur);
    __syncthreads();  // reads of `cur` done before it is refilled at kt+2
  }
  // peeled last stage
  asm volatile("s_wait_asynccnt 0x0" ::: "memory");
  __syncthreads();
  compute_stage((KT - 1) & 1);

  const int l       = t & 31;
  const int colLane = l & 15;
  const int rowOff  = (l < 16) ? 0 : 8;
#pragma unroll
  for (int mi = 0; mi < 2; ++mi) {
#pragma unroll
    for (int ni = 0; ni < 4; ++ni) {
      int gcol = bn0 + wn * 64 + ni * 16 + colLane;
      float bv = bias ? bias[gcol] : 0.0f;
#pragma unroll
      for (int r = 0; r < 8; ++r) {
        int grow = bm0 + wm * 32 + mi * 16 + rowOff + r;
        float v = acc[mi][ni][r] + bv;
        if (outf) outf[(size_t)grow * N + gcol] = v;
        else      outh[(size_t)grow * N + gcol] = (_Float16)v;
      }
    }
  }
}

// ---------------- flash attention: per-(128-query tile, head, seq) block ----------------

#define QTILE 128
#define KCH   32
#define KS    80  // K/V LDS row stride in halves (160B, 16B-aligned)
#define PS    40  // P LDS row stride in halves

__global__ __launch_bounds__(256)
void attn_kernel(const _Float16* __restrict__ Qm, const _Float16* __restrict__ Km,
                 const _Float16* __restrict__ Vm, const long long* __restrict__ seq_len,
                 _Float16* __restrict__ Om) {
  __shared__ _Float16 Kt[KCH * KS];
  __shared__ _Float16 Vt[KCH * KS];
  __shared__ _Float16 Pl[8 * 16 * PS];

  const int t   = threadIdx.x;
  const int wid = t >> 5;
  const int l   = t & 31;
  const int h   = blockIdx.y;
  const int s   = blockIdx.z;

  long long base = 0;
  for (int i = 0; i < s; ++i) base += seq_len[i];
  const int slen = (int)seq_len[s];
  const int q0   = blockIdx.x * QTILE;
  if (q0 >= slen) return;  // uniform across block

  const int qw      = q0 + wid * 16;  // this wave's first query row
  const int colLane = l & 15;
  const int rowOff  = (l < 16) ? 0 : 8;

  // Q fragments (16 queries x 64 dims -> two 16x32 A-frags), rows clamped to slen
  v16h qa0, qa1;
  {
    int rr = qw + colLane;
    if (rr >= slen) rr = slen - 1;
    const _Float16* p = Qm + (size_t)(base + rr) * EMBED + h * HEAD_DIM;
    const int koff = (l < 16) ? 0 : 8;
#pragma unroll
    for (int e = 0; e < 16; ++e) {
      int k = ((e < 8) ? e : (e + 8)) + koff;
      qa0[e] = p[k];
      qa1[e] = p[k + 32];
    }
  }

  v8f o[4];
#pragma unroll
  for (int d = 0; d < 4; ++d) o[d] = (v8f){0.f, 0.f, 0.f, 0.f, 0.f, 0.f, 0.f, 0.f};
  float mrow[8], lrow[8];
#pragma unroll
  for (int r = 0; r < 8; ++r) { mrow[r] = -3.0e38f; lrow[r] = 0.f; }

  const int kend = min(slen, q0 + QTILE);  // causal upper bound for this block
  const float scale = 0.125f;              // 1/sqrt(64)

  for (int k0 = 0; k0 < kend; k0 += KCH) {
    __syncthreads();  // previous chunk's LDS reads complete before refill
    {  // async K/V chunk stage: 32 rows x 64 halves, one b128 each for K and V per thread
      int row = t >> 3;
      int c   = (t & 7) * 8;
      const _Float16* kp = Km + (size_t)(base + k0 + row) * EMBED + h * HEAD_DIM + c;
      const _Float16* vp = Vm + (size_t)(base + k0 + row) * EMBED + h * HEAD_DIM + c;
      async_ld_lds_b128(kp, &Kt[row * KS + c]);
      async_ld_lds_b128(vp, &Vt[row * KS + c]);
      asm volatile("s_wait_asynccnt 0x0" ::: "memory");
    }
    __syncthreads();

    // S = Q * K^T for 16 queries x 32 keys (two 16-key n-tiles, two 32-dim k-chunks)
    v8f s0 = (v8f){0.f, 0.f, 0.f, 0.f, 0.f, 0.f, 0.f, 0.f};
    v8f s1 = s0;
    s0 = WMMA_F16(qa0, load_b_frag_nk(Kt, KS), s0);
    s0 = WMMA_F16(qa1, load_b_frag_nk(Kt + 32, KS), s0);
    s1 = WMMA_F16(qa0, load_b_frag_nk(Kt + 16 * KS, KS), s1);
    s1 = WMMA_F16(qa1, load_b_frag_nk(Kt + 16 * KS + 32, KS), s1);

    // causal mask + scale, row max
    float mloc[8];
#pragma unroll
    for (int r = 0; r < 8; ++r) {
      int qi = qw + rowOff + r;
      float a = ((k0 + colLane) <= qi)      ? s0[r] * scale : -3.0e38f;
      float b = ((k0 + 16 + colLane) <= qi) ? s1[r] * scale : -3.0e38f;
      s0[r] = a; s1[r] = b;
      mloc[r] = fmaxf(a, b);
    }
#pragma unroll
    for (int off = 1; off < 16; off <<= 1)
#pragma unroll
      for (int r = 0; r < 8; ++r) mloc[r] = fmaxf(mloc[r], __shfl_xor(mloc[r], off, 32));

    // online softmax update
    _Float16* Pw = &Pl[wid * 16 * PS];
#pragma unroll
    for (int r = 0; r < 8; ++r) {
      float mn = fmaxf(mrow[r], mloc[r]);
      float sc = __expf(mrow[r] - mn);
      mrow[r] = mn;
      float p0 = (s0[r] > -1.0e37f) ? __expf(s0[r] - mn) : 0.f;
      float p1 = (s1[r] > -1.0e37f) ? __expf(s1[r] - mn) : 0.f;
      float ps = p0 + p1;
#pragma unroll
      for (int off = 1; off < 16; off <<= 1) ps += __shfl_xor(ps, off, 32);
      lrow[r] = lrow[r] * sc + ps;
#pragma unroll
      for (int d = 0; d < 4; ++d) o[d][r] *= sc;
      int prow = rowOff + r;
      Pw[prow * PS + colLane]      = (_Float16)p0;
      Pw[prow * PS + 16 + colLane] = (_Float16)p1;
    }
    __syncthreads();

    // O += P(16x32) * V(32x64)
    v16h pa = load_a_frag(Pw, PS);
    o[0] = WMMA_F16(pa, load_b_frag_kn(Vt + 0, KS), o[0]);
    o[1] = WMMA_F16(pa, load_b_frag_kn(Vt + 16, KS), o[1]);
    o[2] = WMMA_F16(pa, load_b_frag_kn(Vt + 32, KS), o[2]);
    o[3] = WMMA_F16(pa, load_b_frag_kn(Vt + 48, KS), o[3]);
  }

  // normalize and store f16
  float inv[8];
#pragma unroll
  for (int r = 0; r < 8; ++r) inv[r] = (lrow[r] > 0.f) ? 1.0f / lrow[r] : 0.f;
#pragma unroll
  for (int d = 0; d < 4; ++d)
#pragma unroll
    for (int r = 0; r < 8; ++r) {
      int qi = qw + rowOff + r;
      if (qi < slen)
        Om[(size_t)(base + qi) * EMBED + h * HEAD_DIM + d * 16 + colLane] =
            (_Float16)(o[d][r] * inv[r]);
    }
}

// ---------------- host launcher ----------------

extern "C" void kernel_launch(void* const* d_in, const int* in_sizes, int n_in,
                              void* d_out, int out_size, void* d_ws, size_t ws_size,
                              hipStream_t stream) {
  const float*     hs   = (const float*)d_in[0];
  const long long* slen = (const long long*)d_in[1];
  const float*     Wq   = (const float*)d_in[2];
  const float*     bq   = (const float*)d_in[3];
  const float*     Wk   = (const float*)d_in[4];
  const float*     Wv   = (const float*)d_in[5];
  const float*     bv   = (const float*)d_in[6];
  const float*     Wo   = (const float*)d_in[7];
  const float*     bo   = (const float*)d_in[8];
  float*           out  = (float*)d_out;

  const size_t HID_B = (size_t)TOTAL * EMBED * sizeof(_Float16);  // 16 MB
  const size_t W_B   = (size_t)EMBED * EMBED * sizeof(_Float16);  //  2 MB
  char* ws = (char*)d_ws;
  _Float16* hid_h = (_Float16*)(ws);
  _Float16* wq_h  = (_Float16*)(ws + HID_B);
  _Float16* wk_h  = (_Float16*)(ws + HID_B + 1 * W_B);
  _Float16* wv_h  = (_Float16*)(ws + HID_B + 2 * W_B);
  _Float16* wo_h  = (_Float16*)(ws + HID_B + 3 * W_B);
  _Float16* q_h   = (_Float16*)(ws + HID_B + 4 * W_B);
  _Float16* k_h   = (_Float16*)(ws + 2 * HID_B + 4 * W_B);
  _Float16* v_h   = (_Float16*)(ws + 3 * HID_B + 4 * W_B);
  _Float16* att_h = (_Float16*)(ws + 4 * HID_B + 4 * W_B);
  // total workspace use: 5*16MB + 4*2MB = 88 MB

  const int NHID = TOTAL * EMBED;
  const int NW   = EMBED * EMBED;
  cvt_f32_to_f16<<<(NHID + 255) / 256, 256, 0, stream>>>(hs, hid_h, NHID);
  cvt_f32_to_f16<<<(NW + 255) / 256, 256, 0, stream>>>(Wq, wq_h, NW);
  cvt_f32_to_f16<<<(NW + 255) / 256, 256, 0, stream>>>(Wk, wk_h, NW);
  cvt_f32_to_f16<<<(NW + 255) / 256, 256, 0, stream>>>(Wv, wv_h, NW);
  cvt_f32_to_f16<<<(NW + 255) / 256, 256, 0, stream>>>(Wo, wo_h, NW);

  dim3 ggrid(TOTAL / BM, EMBED / BN);
  gemm_nt_kernel<<<ggrid, 256, 0, stream>>>(hid_h, wq_h, bq, q_h, nullptr, TOTAL, EMBED, EMBED);
  gemm_nt_kernel<<<ggrid, 256, 0, stream>>>(hid_h, wk_h, nullptr, k_h, nullptr, TOTAL, EMBED, EMBED);
  gemm_nt_kernel<<<ggrid, 256, 0, stream>>>(hid_h, wv_h, bv, v_h, nullptr, TOTAL, EMBED, EMBED);

  dim3 agrid(SEQ_MAX / QTILE, HEADS, NUM_SEQS);
  attn_kernel<<<agrid, 256, 0, stream>>>(q_h, k_h, v_h, slen, att_h);

  gemm_nt_kernel<<<ggrid, 256, 0, stream>>>(att_h, wo_h, bo, nullptr, out, TOTAL, EMBED, EMBED);
}
